// Experts_30502857736496
// MI455X (gfx1250) — matrix operand order repaired and checked by reading, not compiled, not run on hardware
//
#include <hip/hip_runtime.h>
#include <hip/hip_bf16.h>
#include <math.h>

typedef __bf16 bf16_t;
typedef __attribute__((ext_vector_type(16))) __bf16 v16bf;
typedef __attribute__((ext_vector_type(8)))  __bf16 v8bf;
typedef __attribute__((ext_vector_type(8)))  float  v8f;

#define E_ 16
#define N_ 8192
#define D_ 1024
#define H_ 8
#define BLOCK_ 256
#define NWAVES_ 8
#define TOKS_PER_WAVE_ 16
#define TOKS_PER_BLOCK_ (NWAVES_ * TOKS_PER_WAVE_)   // 128

__device__ __forceinline__ float gelu_exact(float v) {
    // torch nn.GELU default == exact erf GELU
    return 0.5f * v * (1.0f + erff(v * 0.70710678118654752440f));
}

__global__ __launch_bounds__(BLOCK_)
void experts_mlp_wmma(const bf16_t* __restrict__ x,
                      const bf16_t* __restrict__ w1,
                      const bf16_t* __restrict__ w2,
                      bf16_t* __restrict__ out) {
    // WMMA-friendly weight layouts staged in LDS (weights are tiny)
    __shared__ alignas(32) bf16_t w1t[16 * D_];   // [hPad=16][D], rows 8..15 = 0
    __shared__ alignas(16) bf16_t w2t[D_ * H_];   // [D][H]  (== w2 transposed)

    const int tid = threadIdx.x;
    const int e   = blockIdx.y;

    const bf16_t* w1e = w1 + (size_t)e * D_ * H_;   // w1[e][d][h]
    const bf16_t* w2e = w2 + (size_t)e * H_ * D_;   // w2[e][h][d]

    // Stage + transpose weights (coalesced global reads)
    for (int i = tid; i < D_ * H_; i += BLOCK_) {   // i = d*H + h
        int d = i >> 3, h = i & 7;
        w1t[h * D_ + d]       = w1e[i];
        w1t[(8 + h) * D_ + d] = (bf16_t)0.0f;       // padding rows h = 8..15
    }
    for (int i = tid; i < H_ * D_; i += BLOCK_) {   // i = h*D + d
        int h = i >> 10, d = i & (D_ - 1);
        w2t[d * H_ + h] = w2e[i];
    }
    __syncthreads();

    const int lane = tid & 31;
    const int wv   = tid >> 5;
    const int col  = lane & 15;      // token index within the 16-token tile
    const int hi   = lane >> 4;      // half-wave (selects K / M sub-range)

    const size_t tokBase = (size_t)blockIdx.x * TOKS_PER_BLOCK_ + (size_t)wv * TOKS_PER_WAVE_;
    const bf16_t* xrow = x   + ((size_t)e * N_ + tokBase + col) * D_;
    bf16_t*       orow = out + ((size_t)e * N_ + tokBase + col) * D_;
    const bf16_t* w1row = &w1t[col * D_];

    // ------- GEMM1 (transposed): hiddenT[16hPad,16tok] = w1T[16,D] * xT[D,16] -------
    // A 16x32: lane row m=h=col; elems e = K hi*16+e  (two 16B ds loads)
    // B 32x16: lane col n=token=col; elems e = K hi*16+e  (one 32B NT global load)
    v8f acc = {0.f,0.f,0.f,0.f,0.f,0.f,0.f,0.f};
    #pragma unroll 8
    for (int kc = 0; kc < D_; kc += 32) {
        __builtin_prefetch(xrow + kc + 256, 0, 1);   // global_prefetch_b8 stream-ahead
        v8bf a_lo = *reinterpret_cast<const v8bf*>(w1row + kc + hi * 16);
        v8bf a_hi = *reinterpret_cast<const v8bf*>(w1row + kc + hi * 16 + 8);
        v16bf a   = __builtin_shufflevector(a_lo, a_hi,
                        0,1,2,3,4,5,6,7,8,9,10,11,12,13,14,15);
        v16bf b   = __builtin_nontemporal_load(
                        reinterpret_cast<const v16bf*>(xrow + kc + hi * 16));
        acc = __builtin_amdgcn_wmma_f32_16x16x32_bf16(false, a, false, b,
                                                      (short)0, acc, false, false);
    }

    // ------- exact GELU; acc IS already GEMM2's B fragment (hiddenT) -------
    // Lane holds hidden[token=col][h = v+hi*8]; h>=8 accumulators are exactly 0
    // (w1t padded rows) and gelu(0)=0, so B is zero for ALL logical K >= 8.
    v16bf hv;
    #pragma unroll
    for (int q = 0; q < 8; q++)  hv[q] = (bf16_t)gelu_exact(acc[q]);
    #pragma unroll
    for (int q = 8; q < 16; q++) hv[q] = (bf16_t)0.0f;

    // ------- GEMM2 (transposed): outT[16d,16tok] = w2T[D,8pad32] * hiddenT -------
    // Because B (hv) is exactly zero for K>=8, A needs NO lane-dependent masking:
    // hi=1 lanes' elements 0-7 (K=8-15) hold finite w2 values that multiply by 0.
    // Elements 8-15 (K>=16) are compile-time zeros via shuffle.
    // Result: lane holds out[token=col][d0 + hi*8 + v], v=0..7 -> one NT b128 store.
    const v8f  zeroC = {0.f,0.f,0.f,0.f,0.f,0.f,0.f,0.f};
    const v8bf z8    = {(bf16_t)0.f,(bf16_t)0.f,(bf16_t)0.f,(bf16_t)0.f,
                        (bf16_t)0.f,(bf16_t)0.f,(bf16_t)0.f,(bf16_t)0.f};
    #pragma unroll 4
    for (int nc = 0; nc < D_ / 16; nc++) {
        const int d0 = nc * 16;
        v8bf wvals = *reinterpret_cast<const v8bf*>(&w2t[(d0 + col) * H_]);
        v16bf a2   = __builtin_shufflevector(wvals, z8,
                         0,1,2,3,4,5,6,7,8,9,10,11,12,13,14,15);
        v8f o = __builtin_amdgcn_wmma_f32_16x16x32_bf16(false, a2, false, hv,
                                                        (short)0, zeroC, false, false);
        v8bf ov;
        #pragma unroll
        for (int v = 0; v < 8; v++) ov[v] = (bf16_t)o[v];
        __builtin_nontemporal_store(ov,
            reinterpret_cast<v8bf*>(orow + d0 + hi * 8));   // global_store_b128 TH_NT
    }
}

extern "C" void kernel_launch(void* const* d_in, const int* in_sizes, int n_in,
                              void* d_out, int out_size, void* d_ws, size_t ws_size,
                              hipStream_t stream) {
    (void)in_sizes; (void)n_in; (void)out_size; (void)d_ws; (void)ws_size;
    const bf16_t* x  = reinterpret_cast<const bf16_t*>(d_in[0]);
    const bf16_t* w1 = reinterpret_cast<const bf16_t*>(d_in[1]);
    const bf16_t* w2 = reinterpret_cast<const bf16_t*>(d_in[2]);
    bf16_t* out = reinterpret_cast<bf16_t*>(d_out);
    dim3 grid(N_ / TOKS_PER_BLOCK_, E_);
    dim3 block(BLOCK_);
    hipLaunchKernelGGL(experts_mlp_wmma, grid, block, 0, stream, x, w1, w2, out);
}
